// SslTestNet_79207786872990
// MI455X (gfx1250) — matrix operand order repaired and checked by reading, not compiled, not run on hardware
//
#include <hip/hip_runtime.h>
#include <hip/hip_bf16.h>
#include <stdint.h>

// Problem constants (from reference)
#define BATCH 2048
#define TLEN 512
#define CI 48
#define CE 16
#define EMB 64
#define CSD 256
#define NS 150
#define KI (NS * CI)   // 7200
#define KE (NS * CE)   // 2400
#define KT (KI + KE)   // 9600

typedef __attribute__((ext_vector_type(16))) __bf16 v16bf;
typedef __attribute__((ext_vector_type(8)))  __bf16 v8bf;
typedef __attribute__((ext_vector_type(8)))  float  v8f;

// -------------------------------------------------------------------------
// Kernel 1: build fused weight, stored TRANSPOSED: W2T[o][k] (bf16),
//   o in [0,256), k in [0,9600)
//   k < 7200: t=k/48, c=k%48 -> sum_e w_emb_imu[c,e] * w_proj[t*64+e, o]
//   else    : t,c from EMG   -> sum_e w_emb_emg[c,e] * w_proj[t*64+e, o]
// Transposed layout makes the GEMM's B-tile K-contiguous (no LDS transpose).
// -------------------------------------------------------------------------
__global__ void prep_w2_kernel(const float* __restrict__ w_emb_imu,
                               const float* __restrict__ w_emb_emg,
                               const float* __restrict__ w_proj,
                               __bf16* __restrict__ W2T) {
    const int k = blockIdx.x;
    const int o = threadIdx.x;
    const float* we;
    int t, c;
    if (k < KI) { t = k / CI; c = k % CI; we = w_emb_imu + c * EMB; }
    else        { int kk = k - KI; t = kk / CE; c = kk % CE; we = w_emb_emg + c * EMB; }
    float acc = 0.f;
    const float* wp = w_proj + (size_t)t * EMB * CSD + o;
#pragma unroll 8
    for (int e = 0; e < EMB; ++e)
        acc = fmaf(we[e], wp[(size_t)e * CSD], acc);
    W2T[(size_t)o * KT + k] = (__bf16)acc;   // transposed store (one-time)
}

// -------------------------------------------------------------------------
// Kernel 2: folded constant per output: const[p][o] = b_proj[o] +
//           sum_{t,e} b_emb[e] * w_proj[t*64+e, o]
// -------------------------------------------------------------------------
__global__ void prep_const_kernel(const float* __restrict__ b_emb_imu,
                                  const float* __restrict__ b_emb_emg,
                                  const float* __restrict__ b_proj,
                                  const float* __restrict__ w_proj,
                                  float* __restrict__ constv) {
    const int p = blockIdx.x;
    const int o = threadIdx.x;
    const float* be = (p == 0) ? b_emb_imu : b_emb_emg;
    float acc = b_proj[o];
    for (int t = 0; t < NS; ++t) {
        const float* wp = w_proj + (size_t)t * EMB * CSD + o;
#pragma unroll 8
        for (int e = 0; e < EMB; ++e)
            acc = fmaf(be[e], wp[(size_t)e * CSD], acc);
    }
    constv[p * CSD + o] = acc;
}

// -------------------------------------------------------------------------
// Kernel 3: ragged linear interpolation (gather) of raw inputs -> bf16
//   Xq[b][t*48+c]           = interp(x_imu[b,:,c])  (k < 7200)
//   Xq[b][7200 + t*16 + c]  = interp(x_emg[b,:,c])
// Matches torch F.interpolate(linear, align_corners=False) on x[:len].
// -------------------------------------------------------------------------
__global__ void interp_kernel(const float* __restrict__ x_imu,
                              const float* __restrict__ x_emg,
                              const int* __restrict__ lens,
                              __bf16* __restrict__ Xq) {
    const int b = blockIdx.x;
    const int len = lens[b];
    const float scale = (float)len / (float)NS;
    const float* xi = x_imu + (size_t)b * TLEN * CI;
    const float* xe = x_emg + (size_t)b * TLEN * CE;
    __bf16* dst = Xq + (size_t)b * KT;

    for (int j = threadIdx.x; j < KT; j += blockDim.x) {
        const int t = j >> 6;        // 64 channels per timestep (48 imu + 16 emg)
        const int ch = j & 63;
        float src = ((float)t + 0.5f) * scale - 0.5f;
        src = fmaxf(src, 0.f);
        int i0 = (int)src;           // src >= 0 -> trunc == floor
        if (i0 > len - 1) i0 = len - 1;
        int i1 = min(i0 + 1, len - 1);
        const float w = src - (float)i0;
        float v;
        int d;
        if (ch < CI) {
            v = xi[i0 * CI + ch] * (1.f - w) + xi[i1 * CI + ch] * w;
            d = t * CI + ch;
        } else {
            const int c = ch - CI;
            v = xe[i0 * CE + c] * (1.f - w) + xe[i1 * CE + c] * w;
            d = KI + t * CE + c;
        }
        dst[d] = (__bf16)v;
    }
}

// -------------------------------------------------------------------------
// Kernel 4: bf16 WMMA GEMM, double-buffered LDS.
//   p = blockIdx.z selects output (0: imu, K rows [0,7200); 1: emg, [7200,9600))
//   out[p][b][o] = sum_k Xq[b][Kbase+k] * W2T[o][Kbase+k] + constv[p][o]
// Workgroup: 256 threads (8 waves), tile 64(M) x 128(N), K-step 32.
// Each wave: 32x32 = 2x2 grid of v_wmma_f32_16x16x32_bf16 accumulators.
// Both A and B tiles are K-contiguous in global AND in LDS, so every LDS
// access in the hot loop is a b128 op; one barrier per K-step.
// -------------------------------------------------------------------------
__global__ void __launch_bounds__(256)
gemm_wmma_kernel(const __bf16* __restrict__ Xq,
                 const __bf16* __restrict__ WT,
                 const float* __restrict__ constv,
                 float* __restrict__ out) {
    __shared__ __bf16 As[2][64][40];    // [buf][m][k]  (+8 pad)
    __shared__ __bf16 Bs[2][128][40];   // [buf][n][k]  (+8 pad)

    const int tid = threadIdx.x;
    const int p = blockIdx.z;
    const int Kbase = (p == 0) ? 0 : KI;
    const int Ksteps = (p == 0) ? (KI / 32) : (KE / 32);
    const int m0 = blockIdx.x * 64;
    const int n0 = blockIdx.y * 128;

    const int wid = tid >> 5;
    const int lane = tid & 31;
    const int half = lane >> 4;
    const int l16 = lane & 15;
    const int wave_m = (wid & 1) * 32;
    const int wave_n = (wid >> 1) * 32;

    // per-thread tile-load coordinates (K-step = 32 = four 8-elt chunks/row)
    const int rowA = tid >> 2;                 // 64 rows, 4 chunks each
    const int colA = (tid & 3) * 8;
    const __bf16* aSrc = Xq + (size_t)(m0 + rowA) * KT + Kbase + colA;
    const int rowB0 = tid >> 2;                // chunks c=tid and c=tid+256
    const int rowB1 = (tid + 256) >> 2;        // -> n rows 0..127
    const int colB = (tid & 3) * 8;
    const __bf16* bSrc0 = WT + (size_t)(n0 + rowB0) * KT + Kbase + colB;
    const __bf16* bSrc1 = WT + (size_t)(n0 + rowB1) * KT + Kbase + colB;

    v8f acc[2][2];
#pragma unroll
    for (int mi = 0; mi < 2; ++mi)
#pragma unroll
        for (int ni = 0; ni < 2; ++ni)
            acc[mi][ni] = (v8f)(0.f);

    // ---- prologue: stage tile 0 into buffer 0
    {
        v8bf a0 = *(const v8bf*)(aSrc);
        v8bf b0 = *(const v8bf*)(bSrc0);
        v8bf b1 = *(const v8bf*)(bSrc1);
        *(v8bf*)&As[0][rowA][colA] = a0;
        *(v8bf*)&Bs[0][rowB0][colB] = b0;
        *(v8bf*)&Bs[0][rowB1][colB] = b1;
    }
    __syncthreads();

    for (int ks = 0; ks < Ksteps; ++ks) {
        const int cur = ks & 1;
        const int nxt = cur ^ 1;
        const bool more = (ks + 1) < Ksteps;

        // ---- issue next tile's global loads early (overlap with WMMA)
        v8bf aR, bR0, bR1;
        if (more) {
            const int kb = (ks + 1) * 32;
            aR  = *(const v8bf*)(aSrc + kb);
            bR0 = *(const v8bf*)(bSrc0 + kb);
            bR1 = *(const v8bf*)(bSrc1 + kb);
        }

        // ---- fragments per CDNA5 layouts, then 4 WMMAs
        v16bf af[2], bfr[2];
#pragma unroll
        for (int mi = 0; mi < 2; ++mi) {
            const __bf16* ar = &As[cur][wave_m + mi * 16 + l16][0];
            v8bf lo = *(const v8bf*)(ar + half * 8);         // K = half*8 + [0,8)
            v8bf hi = *(const v8bf*)(ar + 16 + half * 8);    // K = 16 + half*8 + [0,8)
            af[mi] = __builtin_shufflevector(lo, hi, 0, 1, 2, 3, 4, 5, 6, 7,
                                             8, 9, 10, 11, 12, 13, 14, 15);
        }
#pragma unroll
        for (int ni = 0; ni < 2; ++ni) {
            const __bf16* br = &Bs[cur][wave_n + ni * 16 + l16][0];
            v8bf lo = *(const v8bf*)(br + half * 8);         // K = half*8 + [0,8)
            v8bf hi = *(const v8bf*)(br + 16 + half * 8);    // K = 16 + half*8 + [0,8)
            bfr[ni] = __builtin_shufflevector(lo, hi, 0, 1, 2, 3, 4, 5, 6, 7,
                                              8, 9, 10, 11, 12, 13, 14, 15);
        }
#pragma unroll
        for (int mi = 0; mi < 2; ++mi)
#pragma unroll
            for (int ni = 0; ni < 2; ++ni)
                acc[mi][ni] = __builtin_amdgcn_wmma_f32_16x16x32_bf16(
                    false, af[mi], false, bfr[ni], (short)0, acc[mi][ni],
                    false, false);

        // ---- stage next tile into the other buffer
        if (more) {
            *(v8bf*)&As[nxt][rowA][colA] = aR;
            *(v8bf*)&Bs[nxt][rowB0][colB] = bR0;
            *(v8bf*)&Bs[nxt][rowB1][colB] = bR1;
        }
        __syncthreads();
    }

    // ---- epilogue: D layout M = 8*half + r, N = l16
    float* outp = out + (size_t)p * BATCH * CSD;
#pragma unroll
    for (int mi = 0; mi < 2; ++mi) {
#pragma unroll
        for (int ni = 0; ni < 2; ++ni) {
            const int nIdx = n0 + wave_n + ni * 16 + l16;
            const float cv = constv[p * CSD + nIdx];
#pragma unroll
            for (int r = 0; r < 8; ++r) {
                const int mIdx = m0 + wave_m + mi * 16 + half * 8 + r;
                outp[(size_t)mIdx * CSD + nIdx] = acc[mi][ni][r] + cv;
            }
        }
    }
}

// -------------------------------------------------------------------------
// Launch: inputs (setup_inputs order):
//   0 x_imu f32[2048,512,48], 1 x_emg f32[2048,512,16], 2 lens i32[2048],
//   3 w_emb_imu f32[48,64], 4 b_emb_imu f32[64], 5 w_emb_emg f32[16,64],
//   6 b_emb_emg f32[64], 7 w_proj f32[9600,256], 8 b_proj f32[256]
// Output: out_imu f32[2048,256] ++ out_emg f32[2048,256]
// Workspace: Xq bf16[2048*9600] (39.3MB) | W2T bf16[256*9600] (4.9MB) |
//            constv f32[512]  => ~44.2 MB total
// -------------------------------------------------------------------------
extern "C" void kernel_launch(void* const* d_in, const int* in_sizes, int n_in,
                              void* d_out, int out_size, void* d_ws, size_t ws_size,
                              hipStream_t stream) {
    const float* x_imu     = (const float*)d_in[0];
    const float* x_emg     = (const float*)d_in[1];
    const int*   lens      = (const int*)d_in[2];
    const float* w_emb_imu = (const float*)d_in[3];
    const float* b_emb_imu = (const float*)d_in[4];
    const float* w_emb_emg = (const float*)d_in[5];
    const float* b_emb_emg = (const float*)d_in[6];
    const float* w_proj    = (const float*)d_in[7];
    const float* b_proj    = (const float*)d_in[8];

    const size_t XQ_BYTES = (size_t)BATCH * KT * sizeof(__bf16);   // 39,321,600
    const size_t W2_BYTES = (size_t)KT * CSD * sizeof(__bf16);     //  4,915,200

    char* ws = (char*)d_ws;
    __bf16* Xq     = (__bf16*)ws;
    __bf16* W2T    = (__bf16*)(ws + XQ_BYTES);
    float*  constv = (float*)(ws + XQ_BYTES + W2_BYTES);

    prep_w2_kernel<<<KT, CSD, 0, stream>>>(w_emb_imu, w_emb_emg, w_proj, W2T);
    prep_const_kernel<<<2, CSD, 0, stream>>>(b_emb_imu, b_emb_emg, b_proj, w_proj, constv);
    interp_kernel<<<BATCH, 256, 0, stream>>>(x_imu, x_emg, lens, Xq);

    dim3 grid(BATCH / 64, CSD / 128, 2);
    gemm_wmma_kernel<<<grid, 256, 0, stream>>>(Xq, W2T, constv, (float*)d_out);
}
